// ResidualColorVoxelDNN_72919954752307
// MI455X (gfx1250) — compile-verified, matrix operand order checked by reading, not run on hardware
//
#include <hip/hip_runtime.h>
#include <hip/hip_bf16.h>

typedef __bf16 bf16_t;
typedef __attribute__((ext_vector_type(16))) __bf16 v16bf;
typedef __attribute__((ext_vector_type(8)))  float  v8f;
typedef int v4i_async __attribute__((vector_size(16)));   // builtin's pointee type

#define NVOX 32768
#define GRID 32

#if defined(__has_builtin)
#if __has_builtin(__builtin_amdgcn_global_load_async_to_lds_b128)
#define HAVE_ASYNC_LDS 1
#endif
#endif

__device__ __forceinline__ float eluf(float v) { return v > 0.0f ? v : (__expf(v) - 1.0f); }

__device__ __forceinline__ void async_stage_fence() {
#ifdef HAVE_ASYNC_LDS
#if __has_builtin(__builtin_amdgcn_s_wait_asynccnt)
  __builtin_amdgcn_s_wait_asynccnt(0);
#else
  asm volatile("s_wait_asynccnt 0x0" ::: "memory");
#endif
#endif
}

// ---------------------------------------------------------------------------
// Weight packing: fp32 [taps, Cin, Cout] (kept-tap prefix) -> bf16 WMMA B-frags.
// Fragment (t, kc, nc) = 512 bf16, laid out [lane 0..31][e 0..15]:
//   lane serves column N = nc*16 + (lane & 15)
//   element e holds     K = kc*32 + ((e>>3)<<4) + ((lane>>4)<<3) + (e&7)
// ---------------------------------------------------------------------------
__global__ void pack_weights_kernel(const float* __restrict__ w, bf16_t* __restrict__ wp,
                                    int total, int cin, int cout, int lg_ntc, int lg_nkc) {
  int i = blockIdx.x * blockDim.x + threadIdx.x;
  if (i >= total) return;
  int e    = i & 15;
  int lane = (i >> 4) & 31;
  int frag = i >> 9;
  int nc   = frag & ((1 << lg_ntc) - 1); frag >>= lg_ntc;
  int kc   = frag & ((1 << lg_nkc) - 1);
  int t    = frag >> lg_nkc;
  int K    = ((e >> 3) << 4) + ((lane >> 4) << 3) + (e & 7);
  int kin  = (kc << 5) + K;
  int n    = (nc << 4) + (lane & 15);
  wp[i] = (bf16_t)w[(t * cin + kin) * cout + n];
}

__global__ void zero_pad_kernel(bf16_t* __restrict__ z) {
  z[threadIdx.x] = (bf16_t)0.0f;   // 128 threads -> 256B zero pad
}

// ---------------------------------------------------------------------------
// Implicit-GEMM masked 3D conv via v_wmma_f32_16x16x32_bf16.
// Workgroup = 8 waves = 128-voxel x (16*NF)-Cout macro-tile. All 8 waves share
// the B fragments, staged whole-tap into dynamic LDS (double-buffered, one
// barrier per tap), via async global->LDS loads when available. Thread 0
// pre-decodes the z-valid tap list (block-uniform) with carry arithmetic ->
// zero runtime division in the hot loop. y/x SAME-padding is branchless via a
// zeroed pad page.
// ---------------------------------------------------------------------------
template <int NF>
__global__ __launch_bounds__(256) void conv3d_wmma_kernel(
    const bf16_t* __restrict__ act,
    const bf16_t* __restrict__ wp,
    const unsigned char* __restrict__ occ,
    const bf16_t* __restrict__ resid,
    bf16_t* __restrict__ outb,
    float* __restrict__ outf,
    const bf16_t* __restrict__ zpad,
    int cin, int cout, int ntaps, int ksz, int lg_ngrp) {
  extern __shared__ __align__(16) bf16_t lsb[];   // (1 or 2) x nkc*NF*1KB
  __shared__ int vtaps[128];
  __shared__ int nvt_s;

  const int tid    = threadIdx.x;
  const int lane   = tid & 31;
  const int wave   = tid >> 5;
  const int ntiles = cout >> 4;
  const int mblk   = blockIdx.x >> lg_ngrp;
  const int ntg    = blockIdx.x & ((1 << lg_ngrp) - 1);

  const int vbase = mblk * 128 + wave * 16;  // this wave's 16 voxels (one x-run)
  const int vz    = vbase >> 10;             // uniform across the block
  const int vy    = (vbase >> 5) & 31;
  const int vxb   = vbase & 31;              // 0 or 16
  const int mrow  = lane & 15;
  const int r0    = (lane >> 4) << 3;        // K sub-offset: 0 (lanes<16) or 8
  const int kh    = ksz >> 1;
  const int nkc   = cin >> 5;
  const int half  = nkc * NF * 512;          // elements per LDS buffer

  // decode taps once (no division): packed {dx+kh, dy+kh, dz+kh, t}
  if (tid == 0) {
    int c = 0, dz = -kh, dy = -kh, dx = -kh;
    for (int t = 0; t < ntaps; ++t) {
      if ((unsigned)(vz + dz) < GRID)
        vtaps[c++] = t | ((dz + kh) << 8) | ((dy + kh) << 16) | ((dx + kh) << 24);
      if (++dx > kh) { dx = -kh; if (++dy > kh) { dy = -kh; ++dz; } }
    }
    nvt_s = c;
  }

  auto stage = [&](int pk, int pb) {   // copy all B frags of one tap into LDS
    const int t = pk & 255;
    if (tid < NF * 64) {
      for (int kc = 0; kc < nkc; ++kc) {
        const uint4* src =
            (const uint4*)(wp + ((size_t)(t * nkc + kc) * ntiles + ntg * NF) * 512) + tid;
        uint4* dst = ((uint4*)(lsb + pb + kc * NF * 512)) + tid;
#ifdef HAVE_ASYNC_LDS
        __builtin_amdgcn_global_load_async_to_lds_b128(
            (v4i_async*)src, (v4i_async*)dst, 0, 0);
#else
        *dst = *src;
#endif
      }
    }
  };

  const v8f vzero = {};
  v8f acc[NF];
#pragma unroll
  for (int nf = 0; nf < NF; ++nf) acc[nf] = vzero;

  __syncthreads();                    // vtaps published
  const int NVT = nvt_s;
  if (NVT > 0) stage(vtaps[0], 0);
  int pb = 0;
  for (int i = 0; i < NVT; ++i) {
    async_stage_fence();              // async LDS writes landed (no-op otherwise)
    __syncthreads();                  // lsb[pb] ready; other half free
    if (i + 1 < NVT) stage(vtaps[i + 1], pb ^ half);

    const int pk = vtaps[i];
    const int dz = ((pk >> 8)  & 15) - kh;
    const int dy = ((pk >> 16) & 15) - kh;
    const int dx = ((pk >> 24) & 15) - kh;
    const int z = vz + dz;            // valid by construction
    const int y = vy + dy;
    const int x = vxb + mrow + dx;
    const bool av = ((unsigned)y < GRID) && ((unsigned)x < GRID);
    const int srcv = (z << 10) + (y << 5) + x;
    const bf16_t* ab = av ? (act + srcv * cin + r0) : zpad;
    const int astep = av ? 32 : 0;
    const bf16_t* lb = lsb + pb + lane * 16;

    for (int kc = 0; kc < nkc; ++kc) {
      union { v16bf v; uint4 q[2]; } a;
      a.q[0] = ((const uint4*)ab)[0];   // K = r0 .. r0+7
      a.q[1] = ((const uint4*)ab)[2];   // K = r0+16 .. r0+23
      ab += astep;
#pragma unroll
      for (int nf = 0; nf < NF; ++nf) {
        union { v16bf v; uint4 q[2]; } b;
        const uint4* lp = (const uint4*)(lb + nf * 512);
        b.q[0] = lp[0];
        b.q[1] = lp[1];
        acc[nf] = __builtin_amdgcn_wmma_f32_16x16x32_bf16(false, a.v, false, b.v,
                                                          (short)0, acc[nf], false, false);
      }
      lb += NF * 512;
    }
    pb ^= half;
  }

  // C/D layout: VGPR r, lanes 0-15 -> M=r, N=lane; lanes 16-31 -> M=r+8, N=lane-16
  const int mo = (lane >> 4) << 3;
#pragma unroll
  for (int nf = 0; nf < NF; ++nf) {
    const int n = ((ntg * NF + nf) << 4) + (lane & 15);
#pragma unroll
    for (int r = 0; r < 8; ++r) {
      const int vox = vbase + mo + r;
      float o = acc[nf][r] * (occ[vox] ? 1.0f : 0.0f);
      if (resid) o += (float)resid[vox * cout + n];
      if (outb)  outb[vox * cout + n] = (bf16_t)o;
      if (outf)  outf[vox * cout + n] = o;
    }
  }
}

// ---------------------------------------------------------------------------
// Cin==1 head convs (7^3 fixed, tiny FLOP share): scalar FMA, no divisions.
// ---------------------------------------------------------------------------
__global__ void conv_cin1_kernel(const float* __restrict__ x, const float* __restrict__ w,
                                 const unsigned char* __restrict__ occ,
                                 bf16_t* __restrict__ out, int lg_cout, int ntaps) {
  int i = blockIdx.x * blockDim.x + threadIdx.x;
  if (i >= (NVOX << lg_cout)) return;
  int cout = 1 << lg_cout;
  int n    = i & (cout - 1);
  int vox  = i >> lg_cout;
  int vz = vox >> 10, vy = (vox >> 5) & 31, vx = vox & 31;
  float acc = 0.0f;
  int dz = -3, dy = -3, dx = -3;                 // ksz = 7
  for (int t = 0; t < ntaps; ++t) {
    int z = vz + dz, yy = vy + dy, xx = vx + dx;
    if ((unsigned)z < GRID && (unsigned)yy < GRID && (unsigned)xx < GRID)
      acc = fmaf(x[(z << 10) + (yy << 5) + xx], w[(t << lg_cout) + n], acc);
    if (++dx > 3) { dx = -3; if (++dy > 3) { dy = -3; ++dz; } }
  }
  out[i] = (bf16_t)(occ[vox] ? acc : 0.0f);
}

// ---------------------------------------------------------------------------
// Elementwise helpers, 8 bf16 per thread (one b128 load/store)
// ---------------------------------------------------------------------------
union BQ { uint4 q; bf16_t b[8]; };

__global__ void elu_kernel(const uint4* __restrict__ s, uint4* __restrict__ d, int n8) {
  int i = blockIdx.x * blockDim.x + threadIdx.x;
  if (i >= n8) return;
  BQ u; u.q = s[i];
#pragma unroll
  for (int j = 0; j < 8; ++j) u.b[j] = (bf16_t)eluf((float)u.b[j]);
  d[i] = u.q;
}
__global__ void fuse_elu_kernel(uint4* __restrict__ h, const uint4* __restrict__ c, int n8) {
  int i = blockIdx.x * blockDim.x + threadIdx.x;
  if (i >= n8) return;
  BQ a, b; a.q = h[i]; b.q = c[i];
#pragma unroll
  for (int j = 0; j < 8; ++j) a.b[j] = (bf16_t)eluf((float)a.b[j] + (float)b.b[j]);
  h[i] = a.q;
}
__global__ void add_kernel(uint4* __restrict__ h, const uint4* __restrict__ c, int n8) {
  int i = blockIdx.x * blockDim.x + threadIdx.x;
  if (i >= n8) return;
  BQ a, b; a.q = h[i]; b.q = c[i];
#pragma unroll
  for (int j = 0; j < 8; ++j) a.b[j] = (bf16_t)((float)a.b[j] + (float)b.b[j]);
  h[i] = a.q;
}

// ---------------------------------------------------------------------------
// Host orchestration
// ---------------------------------------------------------------------------
struct ConvDesc { int idx, ksz, taps, cin, cout; };
// d_in: 0=x 1=y 2=occ 3=head 4-6=res32(a,b,c) 7=m1 8-10=res64 11=m2 12-14=res128
//       15=m3 16-24=res256[0..2] 25-27=tail 28=cc_head 29-31=cc_res 32=chroma_last
static const ConvDesc CV[28] = {
  { 4,1,  1, 64, 32}, { 5,5, 63, 32, 32}, { 6,1,  1, 32, 64},   // res32
  { 7,3, 14, 64,128},                                            // m1 (B mask: 14 taps)
  { 8,1,  1,128, 64}, { 9,5, 63, 64, 64}, {10,1,  1, 64,128},   // res64
  {11,3, 14,128,256},                                            // m2
  {12,1,  1,256,128}, {13,5, 63,128,128}, {14,1,  1,128,256},   // res128
  {15,3, 14,256,512},                                            // m3
  {16,1,  1,512,256}, {17,5, 63,256,256}, {18,1,  1,256,512},   // res256[0]
  {19,1,  1,512,256}, {20,5, 63,256,256}, {21,1,  1,256,512},   // res256[1]
  {22,1,  1,512,256}, {23,5, 63,256,256}, {24,1,  1,256,512},   // res256[2]
  {25,1,  1,512,256}, {26,5, 63,256,256}, {27,1,  1,256,512},   // tail
  {29,1,  1,512,256}, {30,5,125,256,256}, {31,1,  1,256,512},   // cc_res (unmasked b)
  {32,1,  1,512,512},                                            // chroma_last
};

static inline int ilog2i(int v) { int l = 0; while ((1 << l) < v) ++l; return l; }

extern "C" void kernel_launch(void* const* d_in, const int* in_sizes, int n_in,
                              void* d_out, int out_size, void* d_ws, size_t ws_size,
                              hipStream_t stream) {
  (void)in_sizes; (void)n_in; (void)out_size; (void)ws_size;
  const float* x = (const float*)d_in[0];
  const float* y = (const float*)d_in[1];
  const unsigned char* occ = (const unsigned char*)d_in[2];

  char* ws = (char*)d_ws;
  size_t off = 0;
  auto carve = [&](size_t bytes) -> char* {
    char* p = ws + off;
    off += (bytes + 255) & ~(size_t)255;
    return p;
  };

  bf16_t* zpad = (bf16_t*)carve(256);                                 // zero pad page
  bf16_t* bufH = (bf16_t*)carve((size_t)NVOX * 512 * sizeof(bf16_t)); // backbone h
  bf16_t* bufE = (bf16_t*)carve((size_t)NVOX * 512 * sizeof(bf16_t)); // elu scratch
  bf16_t* bufT = (bf16_t*)carve((size_t)NVOX * 512 * sizeof(bf16_t)); // block temp
  bf16_t* bufC = (bf16_t*)carve((size_t)NVOX * 512 * sizeof(bf16_t)); // cc branch

  zero_pad_kernel<<<1, 128, 0, stream>>>(zpad);

  // Pack all WMMA conv weights (fp32 -> bf16 fragments); masked taps skipped
  // because the Minkowski mask keeps a prefix of flattened taps.
  bf16_t* wpk[28];
  for (int i = 0; i < 28; ++i) {
    size_t nelem = (size_t)CV[i].taps * CV[i].cin * CV[i].cout;
    wpk[i] = (bf16_t*)carve(nelem * sizeof(bf16_t));
    int total = (int)nelem;
    pack_weights_kernel<<<(total + 255) / 256, 256, 0, stream>>>(
        (const float*)d_in[CV[i].idx], wpk[i], total, CV[i].cin, CV[i].cout,
        ilog2i(CV[i].cout >> 4), ilog2i(CV[i].cin >> 5));
  }

  auto conv = [&](const bf16_t* in, int ci, int wi, const bf16_t* resid,
                  bf16_t* ob, float* of) {
    int co  = CV[wi].cout;
    int nkc = ci >> 5;
    if (co >= 64) {
      int ngrp = co / 64;
      int blocks = (NVOX / 128) * ngrp;
      size_t shb = (size_t)(CV[wi].taps > 1 ? 2 : 1) * nkc * 4 * 1024;
      conv3d_wmma_kernel<4><<<blocks, 256, shb, stream>>>(
          in, wpk[wi], occ, resid, ob, of, zpad, ci, co, CV[wi].taps, CV[wi].ksz,
          ilog2i(ngrp));
    } else {  // cout == 32
      int blocks = (NVOX / 128);
      size_t shb = (size_t)(CV[wi].taps > 1 ? 2 : 1) * nkc * 2 * 1024;
      conv3d_wmma_kernel<2><<<blocks, 256, shb, stream>>>(
          in, wpk[wi], occ, resid, ob, of, zpad, ci, co, CV[wi].taps, CV[wi].ksz, 0);
    }
  };
  auto elu = [&](const bf16_t* s, bf16_t* d, int n) {
    elu_kernel<<<(n / 8 + 255) / 256, 256, 0, stream>>>((const uint4*)s, (uint4*)d, n / 8);
  };
  // res block: f += conv1x1(elu(conv5(elu(conv1x1(elu(f)))))), occ applied per conv
  auto block = [&](bf16_t* f, int h, int wa) {
    elu(f, bufE, NVOX * 2 * h);
    conv(bufE, 2 * h, wa + 0, nullptr, bufT, nullptr);
    elu(bufT, bufE, NVOX * h);
    conv(bufE, h, wa + 1, nullptr, bufT, nullptr);
    elu(bufT, bufE, NVOX * h);
    conv(bufE, h, wa + 2, f, f, nullptr);
  };

  // Backbone
  conv_cin1_kernel<<<(NVOX * 64 + 255) / 256, 256, 0, stream>>>(
      x, (const float*)d_in[3], occ, bufH, 6, 171);              // head, A-mask: 171 taps
  block(bufH, 32, 0);                                            // res32
  elu(bufH, bufE, NVOX * 64);  conv(bufE, 64, 3, nullptr, bufH, nullptr);   // m1 -> 128
  block(bufH, 64, 4);                                            // res64
  elu(bufH, bufE, NVOX * 128); conv(bufE, 128, 7, nullptr, bufH, nullptr);  // m2 -> 256
  block(bufH, 128, 8);                                           // res128
  elu(bufH, bufE, NVOX * 256); conv(bufE, 256, 11, nullptr, bufH, nullptr); // m3 -> 512
  block(bufH, 256, 12);                                          // res256 x3
  block(bufH, 256, 15);
  block(bufH, 256, 18);

  // Channel-condition branch
  conv_cin1_kernel<<<(NVOX * 512 + 255) / 256, 256, 0, stream>>>(
      y, (const float*)d_in[28], occ, bufC, 9, 343);             // cc_head, unmasked
  elu(bufC, bufC, NVOX * 512);                                   // c = elu(c)
  block(bufC, 256, 24);                                          // cc_res (unmasked 5^3)

  // Fuse + tail + chroma head
  fuse_elu_kernel<<<(NVOX * 512 / 8 + 255) / 256, 256, 0, stream>>>(
      (uint4*)bufH, (const uint4*)bufC, NVOX * 512 / 8);
  block(bufH, 256, 21);                                          // tail
  add_kernel<<<(NVOX * 512 / 8 + 255) / 256, 256, 0, stream>>>(
      (uint4*)bufH, (const uint4*)bufC, NVOX * 512 / 8);
  conv(bufH, 512, 27, nullptr, nullptr, (float*)d_out);          // chroma_last -> fp32
}
